// ConvolutionAttention_50603304681892
// MI455X (gfx1250) — matrix operand order repaired and checked by reading, not compiled, not run on hardware
//
#include <hip/hip_runtime.h>

// ---------------------------------------------------------------------------
// ConvolutionAttention for MI455X (gfx1250, wave32, WMMA bf16 16x16x32)
// B=8, H=W=32, C=256, HEADS=8, DH=64, OC=512, HW=1024
// All GEMM B-operands are stored pre-swizzled in WMMA B-fragment order:
//   tile(32x16 KxN) = 32 lanes x 16 contiguous bf16  (512 elements, 1 KB)
// so every B-fragment load is two coalesced b128 loads.
// Attention K/V chunks are staged into LDS by the Tensor Data Mover
// (tensor_load_to_lds + s_wait_tensorcnt), double-buffered to overlap the
// DMA with WMMA compute; synchronous-copy fallback if the builtin is absent.
// ---------------------------------------------------------------------------

#define BB    8
#define HH    32
#define WW_   32
#define CC    256
#define HEADS 8
#define DHD   64
#define OCC   512
#define HWN   1024
#define EPSV  1e-5f
#define SCALEV 0.125f   // 64^-0.5

typedef __attribute__((ext_vector_type(16))) __bf16          v16bf;
typedef __attribute__((ext_vector_type(16))) unsigned short  v16us;
typedef __attribute__((ext_vector_type(8)))  float           v8f;
typedef __attribute__((ext_vector_type(4)))  unsigned int    v4u32;
typedef __attribute__((ext_vector_type(8)))  int             v8i32;
typedef __attribute__((ext_vector_type(4)))  int             v4i32;

#ifndef __has_builtin
#define __has_builtin(x) 0
#endif
#if __has_builtin(__builtin_amdgcn_tensor_load_to_lds) && \
    __has_builtin(__builtin_amdgcn_s_wait_tensorcnt)
#define HAVE_TDM 1
#else
#define HAVE_TDM 0
#endif

// ---- bf16 bit helpers (round-to-nearest-even) ------------------------------
static __device__ inline unsigned short f2bf(float f) {
    unsigned int u = __float_as_uint(f);
    u += 0x7FFFu + ((u >> 16) & 1u);
    return (unsigned short)(u >> 16);
}

static __device__ inline v8f v8f_zero() {
    v8f z;
#pragma unroll
    for (int i = 0; i < 8; ++i) z[i] = 0.0f;
    return z;
}

// ---- WMMA fragment loaders (contiguous k order, shared by A and B) ------------------
// A: 16x32 (MxK) bf16. lane&15 = M row; lanes 0-15 hold K 0..15,
// lanes 16-31 hold K 16..31. Two 16-byte vector loads per lane.
template <typename PT>
static __device__ inline v16bf load_frag_a(PT base_, int ld) {
    const unsigned short* base = (const unsigned short*)base_;
    int lane = threadIdx.x & 31;
    const unsigned short* p = base + (lane & 15) * ld + ((lane >> 4) << 4);
    union { v16us v; struct { uint4 lo; uint4 hi; } q; } u;
    u.q.lo = *(const uint4*)(p);
    u.q.hi = *(const uint4*)(p + 8);
    return __builtin_bit_cast(v16bf, u.v);
}

// B (pre-swizzled tile): lane's 16 elements are contiguous at tile + lane*16.
template <typename PT>
static __device__ inline v16bf load_frag_b_sw(PT base_) {
    const unsigned short* p =
        (const unsigned short*)base_ + (threadIdx.x & 31) * 16;
    union { v16us v; struct { uint4 lo; uint4 hi; } q; } u;
    u.q.lo = *(const uint4*)(p);
    u.q.hi = *(const uint4*)(p + 8);
    return __builtin_bit_cast(v16bf, u.v);
}

static __device__ inline v8f wmma_bf16(v16bf a, v16bf b, v8f c) {
    return __builtin_amdgcn_wmma_f32_16x16x32_bf16(
        /*neg_a=*/false, a, /*neg_b=*/false, b,
        /*c_mod=*/(short)0, c, /*reuse_a=*/false, /*reuse_b=*/false);
}

#if HAVE_TDM
// ---- TDM: 1-D contiguous tile, global -> LDS (CDNA5 ISA ch.8 D# layout) ----
// data_size=8B; tensor_dim0 = tile_dim0 = bytes/8; stride = tensor_dim0.
static __device__ inline void tdm_load_1d(const void* gsrc, void* lds_dst,
                                          int bytes) {
    unsigned long long ga = (unsigned long long)(size_t)gsrc;
    unsigned int la = (unsigned int)(size_t)lds_dst;  // LDS offset = addr[31:0]
    unsigned int n = (unsigned int)(bytes >> 3);      // 8-byte elements
    v4u32 g0;
    g0[0] = 1u;                                            // count=1 (valid D#)
    g0[1] = la;                                            // lds_addr
    g0[2] = (unsigned int)ga;                              // global_addr[31:0]
    g0[3] = (unsigned int)((ga >> 32) & 0x1FFFFFFu) | (2u << 30);  // type=2
    v8i32 g1;
    g1[0] = (int)(3u << 16);                 // data_size = 8 bytes
    g1[1] = (int)((n & 0xFFFFu) << 16);      // tensor_dim0[15:0]  (bits 63:48)
    g1[2] = (int)((n >> 16) & 0xFFFFu);      // tensor_dim0[31:16] (bits 79:64)
    g1[3] = (int)((n & 0xFFFFu) << 16);      // tile_dim0          (bits 127:112)
    g1[4] = 0;                               // tile_dim1/2 unused (1-D)
    g1[5] = (int)n;                          // tensor_dim0_stride[31:0]
    g1[6] = 0;
    g1[7] = 0;
    v4i32 gz = {0, 0, 0, 0};
#if __clang_major__ >= 23
    v8i32 gz8 = {0, 0, 0, 0, 0, 0, 0, 0};
    __builtin_amdgcn_tensor_load_to_lds(g0, g1, gz, gz, gz8, 0);
#else
    __builtin_amdgcn_tensor_load_to_lds(g0, g1, gz, gz, 0);
#endif
}
#endif

// ---------------------------------------------------------------------------
// Kernel W: weight prep into swizzled B-fragment order (bf16).
// wq/wk/wv: B = pw_w^T (K=c 256, N=oc 512): tiles kt 0..7, nt 0..31.
// wf:       B = ffn_w^T (K=oc 512, N=c 256): tiles kt 0..15, nt 0..15.
// Output element i = ((kt*NT + nt)*32 + lane)*16 + e, with
//   K = kt*32 + ((lane>>4)<<4) + e,  N = nt*16 + (lane&15).
// ---------------------------------------------------------------------------
__global__ __launch_bounds__(256) void prep_weights_kernel(
    const float* __restrict__ qw, const float* __restrict__ kw,
    const float* __restrict__ vw, const float* __restrict__ fw,
    unsigned short* __restrict__ wq, unsigned short* __restrict__ wk,
    unsigned short* __restrict__ wv, unsigned short* __restrict__ wf) {
    int i = blockIdx.x * 256 + threadIdx.x;  // 0..131071
    int e = i & 15, lane = (i >> 4) & 31, tile = i >> 9;
    // pw projections: NT = 32
    {
        int nt = tile & 31, kt = tile >> 5;
        int K = kt * 32 + ((lane >> 4) << 4) + e;   // c
        int N = nt * 16 + (lane & 15);              // oc
        float v = qw[N * CC + K];  wq[i] = f2bf(v); // pw_w[oc][c]
        v = kw[N * CC + K];        wk[i] = f2bf(v);
        v = vw[N * CC + K];        wv[i] = f2bf(v);
    }
    // ffn: NT = 16
    {
        int nt = tile & 15, kt = tile >> 4;
        int K = kt * 32 + ((lane >> 4) << 4) + e;   // oc
        int N = nt * 16 + (lane & 15);              // c
        wf[i] = f2bf(fw[N * OCC + K]);              // ffn_w[c][oc]
    }
}

// ---------------------------------------------------------------------------
// Kernel A: depthwise 3x3 + BN (VALU, into LDS bf16) then pointwise GEMM
// [64 x 512] = ybn[64 x 256] @ wt via WMMA (wt pre-swizzled).
// grid = (HW/64, B), block = 256 (8 waves).
// mode 0: Q  -> [bh][pos][dh]           (A-fragment friendly, pre-scaled)
// mode 1: K  -> swizzled score-B frags: [bh][postile 64][kstep 2][512]
// mode 2: V  -> swizzled PV-B frags:    [bh][keytile 32][dhtile 4][512]
// ---------------------------------------------------------------------------
#define YSTR 264  // 256 + 8 pad (528 bytes/row, 16B-aligned rows)

__global__ __launch_bounds__(256) void sepconv_proj_kernel(
    const float* __restrict__ x,       // [B,1024,256]
    const float* __restrict__ dw_w,    // [256,9]
    const float* __restrict__ dw_b,
    const float* __restrict__ bn_g, const float* __restrict__ bn_b,
    const float* __restrict__ bn_m, const float* __restrict__ bn_v,
    const unsigned short* __restrict__ wt,  // swizzled [8][32][512]
    const float* __restrict__ pw_b,    // [512]
    unsigned short* __restrict__ dst,
    float out_scale, int mode) {
    __shared__ __align__(16) unsigned short ybn[64][YSTR];

    const int b    = blockIdx.y;
    const int pos0 = blockIdx.x * 64;
    const int c    = threadIdx.x;  // one channel per thread

    float w9[9];
#pragma unroll
    for (int i = 0; i < 9; ++i) w9[i] = dw_w[c * 9 + i];
    const float bias = dw_b[c];
    const float inv  = rsqrtf(bn_v[c] + EPSV);
    const float sc   = bn_g[c] * inv;
    const float sh   = bn_b[c] - bn_m[c] * sc;
    const float* xb  = x + (size_t)b * HWN * CC + c;

    for (int pos = 0; pos < 64; ++pos) {
        int p = pos0 + pos;
        int h = p >> 5, w = p & 31;
        float acc = bias;
#pragma unroll
        for (int dy = -1; dy <= 1; ++dy) {
            int hh = h + dy;
            if (hh < 0 || hh >= HH) continue;
#pragma unroll
            for (int dx = -1; dx <= 1; ++dx) {
                int ww = w + dx;
                if (ww < 0 || ww >= WW_) continue;
                acc += xb[(size_t)(hh * WW_ + ww) * CC] * w9[(dy + 1) * 3 + (dx + 1)];
            }
        }
        ybn[pos][c] = f2bf(acc * sc + sh);
    }
    __syncthreads();

    // pointwise GEMM: each wave owns 4 N-tiles across all 4 M-tiles
    const int wave = threadIdx.x >> 5;
    const int lane = threadIdx.x & 31;
    const int ntb  = wave * 4;
    const int n    = lane & 15;
    const int mb   = (lane >> 4) * 8;

    for (int mt = 0; mt < 4; ++mt) {
        v8f acc[4];
#pragma unroll
        for (int j = 0; j < 4; ++j) acc[j] = v8f_zero();
#pragma unroll
        for (int kt = 0; kt < 8; ++kt) {
            v16bf a = load_frag_a(&ybn[mt * 16][kt * 32], YSTR);
#pragma unroll
            for (int j = 0; j < 4; ++j) {
                v16bf bfrag = load_frag_b_sw(wt + (size_t)(kt * 32 + ntb + j) * 512);
                acc[j] = wmma_bf16(a, bfrag, acc[j]);
            }
        }
#pragma unroll
        for (int j = 0; j < 4; ++j) {
            int oc = (ntb + j) * 16 + n;
            float pb = pw_b[oc];
            int head = oc >> 6, dh = oc & 63;
            size_t bh = (size_t)(b * HEADS + head);
#pragma unroll
            for (int r = 0; r < 8; ++r) {
                int gpos = pos0 + mt * 16 + mb + r;
                unsigned short bv = f2bf((acc[j][r] + pb) * out_scale);
                size_t idx;
                if (mode == 0) {
                    idx = (bh * HWN + gpos) * DHD + dh;
                } else if (mode == 1) {
                    // score B-frag: K=dh, N=keypos
                    idx = ((bh * 64 + (size_t)(gpos >> 4)) * 2 + (dh >> 5)) * 512
                        + ((dh >> 4) & 1) * 256 + (gpos & 15) * 16 + (dh & 15);
                } else {
                    // PV B-frag: K=keypos, N=dh
                    idx = ((bh * 32 + (size_t)(gpos >> 5)) * 4 + (dh >> 4)) * 512
                        + ((gpos >> 4) & 1) * 256 + (dh & 15) * 16 + (gpos & 15);
                }
                dst[idx] = bv;
            }
        }
    }
}

// ---------------------------------------------------------------------------
// Kernel B: flash attention. grid=(8, HEADS, B), block=256 (8 waves).
// Per key chunk of 64: the 8 K-fragments + 8 V-fragments (8 KB each,
// contiguous in the swizzled layout) are staged into a double-buffered LDS
// region by the TDM (wave 0 issues tensor_load_to_lds for chunk kc+1 while
// all waves compute on chunk kc; s_wait_tensorcnt + barrier publishes it).
// Online softmax with __shfl_xor reductions over the 16-lane N groups.
// P goes C-layout -> LDS -> A-layout (same-wave LDS ops are in-order).
// ---------------------------------------------------------------------------
#define PSTR 72  // 64 + 8 pad (144 bytes/row)

__global__ __launch_bounds__(256) void attention_kernel(
    const unsigned short* __restrict__ q,    // [bh][1024][64] (pre-scaled)
    const unsigned short* __restrict__ ksw,  // [bh][64][2][512] swizzled
    const unsigned short* __restrict__ vsw,  // [bh][32][4][512] swizzled
    unsigned short* __restrict__ o) {        // [B][1024][512]
    __shared__ __align__(16) unsigned short kvbuf[2][2][4096];  // [buf][K|V]
    __shared__ __align__(16) unsigned short pbuf[8][16][PSTR];

    const int bh   = blockIdx.z * HEADS + blockIdx.y;
    const int wave = threadIdx.x >> 5;
    const int lane = threadIdx.x & 31;
    const int q0   = blockIdx.x * 128 + wave * 16;

    const unsigned short* qb  = q + ((size_t)bh * HWN + q0) * DHD;
    const unsigned short* kc0 = ksw + (size_t)bh * 64 * 2 * 512;
    const unsigned short* vc0 = vsw + (size_t)bh * 32 * 4 * 512;

    const v16bf qa0 = load_frag_a(qb, DHD);       // K = dh 0..31
    const v16bf qa1 = load_frag_a(qb + 32, DHD);  // K = dh 32..63

    v8f ov[4];
#pragma unroll
    for (int j = 0; j < 4; ++j) ov[j] = v8f_zero();
    float mrow[8], lrow[8];
#pragma unroll
    for (int r = 0; r < 8; ++r) { mrow[r] = -1e30f; lrow[r] = 0.0f; }

#if HAVE_TDM
    if (wave == 0) {  // prologue: DMA chunk 0 (TDM ignores EXEC; 1 wave = 1 op)
        tdm_load_1d(kc0, &kvbuf[0][0][0], 8192);
        tdm_load_1d(vc0, &kvbuf[0][1][0], 8192);
    }
#endif

    for (int kc = 0; kc < 16; ++kc) {
        const int cur = kc & 1;
#if HAVE_TDM
        if (wave == 0) __builtin_amdgcn_s_wait_tensorcnt(0);  // chunk kc ready
        __syncthreads();  // publish DMA data; all waves done with buf[cur^1]
        if (wave == 0 && kc + 1 < 16) {  // overlap: DMA chunk kc+1
            tdm_load_1d(kc0 + (size_t)(kc + 1) * 4096, &kvbuf[cur ^ 1][0][0], 8192);
            tdm_load_1d(vc0 + (size_t)(kc + 1) * 4096, &kvbuf[cur ^ 1][1][0], 8192);
        }
#else
        __syncthreads();  // previous iteration done with kvbuf[cur]
        {
            const uint4* gk = (const uint4*)(kc0 + (size_t)kc * 4096);
            const uint4* gv = (const uint4*)(vc0 + (size_t)kc * 4096);
            uint4* lk = (uint4*)&kvbuf[cur][0][0];
            uint4* lv = (uint4*)&kvbuf[cur][1][0];
#pragma unroll
            for (int i = 0; i < 2; ++i) {
                lk[threadIdx.x + i * 256] = gk[threadIdx.x + i * 256];
                lv[threadIdx.x + i * 256] = gv[threadIdx.x + i * 256];
            }
            if (kc + 1 < 16) {  // pull next chunk toward L2 (global_prefetch_b8)
                __builtin_prefetch(gk + 512 + (threadIdx.x & 1) * 256, 0, 0);
                __builtin_prefetch(gv + 512 + (threadIdx.x & 1) * 256, 0, 0);
            }
        }
        __syncthreads();
#endif

        // ---- S = Q @ K^T : 16 queries x 64 keys ----
        v8f s[4];
#pragma unroll
        for (int j = 0; j < 4; ++j) {
            s[j] = v8f_zero();
            v16bf b0 = load_frag_b_sw(&kvbuf[cur][0][(j * 2 + 0) * 512]);  // dh 0..31
            s[j] = wmma_bf16(qa0, b0, s[j]);
            v16bf b1 = load_frag_b_sw(&kvbuf[cur][0][(j * 2 + 1) * 512]);  // dh 32..63
            s[j] = wmma_bf16(qa1, b1, s[j]);
        }

        // ---- online softmax over the 64-key chunk ----
        float pr[4][8];
#pragma unroll
        for (int r = 0; r < 8; ++r) {
            float t = fmaxf(fmaxf(s[0][r], s[1][r]), fmaxf(s[2][r], s[3][r]));
#pragma unroll
            for (int off = 8; off >= 1; off >>= 1) t = fmaxf(t, __shfl_xor(t, off, 32));
            float mn   = fmaxf(mrow[r], t);
            float corr = __expf(mrow[r] - mn);
            mrow[r] = mn;
            float csum = 0.0f;
#pragma unroll
            for (int j = 0; j < 4; ++j) { pr[j][r] = __expf(s[j][r] - mn); csum += pr[j][r]; }
#pragma unroll
            for (int off = 8; off >= 1; off >>= 1) csum += __shfl_xor(csum, off, 32);
            lrow[r] = lrow[r] * corr + csum;
#pragma unroll
            for (int j = 0; j < 4; ++j) ov[j][r] *= corr;
        }

        // ---- P: C-layout -> per-wave LDS -> A-layout (in-order DS ops) ----
        {
            int nn = lane & 15, mb = (lane >> 4) * 8;
#pragma unroll
            for (int j = 0; j < 4; ++j)
#pragma unroll
                for (int r = 0; r < 8; ++r)
                    pbuf[wave][mb + r][j * 16 + nn] = f2bf(pr[j][r]);
        }
        v16bf pa0 = load_frag_a(&pbuf[wave][0][0], PSTR);
        v16bf pa1 = load_frag_a(&pbuf[wave][0][32], PSTR);

        // ---- O += P @ V ----
#pragma unroll
        for (int nt = 0; nt < 4; ++nt) {
            v16bf vb0 = load_frag_b_sw(&kvbuf[cur][1][(0 * 4 + nt) * 512]);  // keys 0..31
            ov[nt] = wmma_bf16(pa0, vb0, ov[nt]);
            v16bf vb1 = load_frag_b_sw(&kvbuf[cur][1][(1 * 4 + nt) * 512]);  // keys 32..63
            ov[nt] = wmma_bf16(pa1, vb1, ov[nt]);
        }
    }

    // epilogue: normalize and store bf16 to [B][1024][512]
    const int nn = lane & 15, mb = (lane >> 4) * 8;
    const int head = bh & (HEADS - 1), b = bh >> 3;
#pragma unroll
    for (int nt = 0; nt < 4; ++nt)
#pragma unroll
        for (int r = 0; r < 8; ++r) {
            int row = q0 + mb + r;
            float val = ov[nt][r] / lrow[r];
            o[((size_t)(b * HWN + row)) * OCC + head * DHD + nt * 16 + nn] = f2bf(val);
        }
}

// ---------------------------------------------------------------------------
// Kernel C: FFN  out[8192,256] = attn[8192,512] @ wf + bias  (fp32 out).
// grid = 128, block = 256 (8 waves); wave owns 2 N-tiles x 4 M-tiles.
// ---------------------------------------------------------------------------
__global__ __launch_bounds__(256) void ffn_kernel(
    const unsigned short* __restrict__ a,   // [8192,512] bf16
    const unsigned short* __restrict__ wf,  // swizzled [16][16][512] bf16
    const float* __restrict__ bias,         // [256]
    float* __restrict__ out) {              // [8192,256] f32
    const int wave = threadIdx.x >> 5;
    const int lane = threadIdx.x & 31;
    const int r0 = blockIdx.x * 64;
    const int nt0 = wave * 2;
    const int nn = lane & 15, mb = (lane >> 4) * 8;

    for (int mt = 0; mt < 4; ++mt) {
        v8f acc0 = v8f_zero(), acc1 = v8f_zero();
#pragma unroll
        for (int kt = 0; kt < 16; ++kt) {
            v16bf av = load_frag_a(a + (size_t)(r0 + mt * 16) * OCC + kt * 32, OCC);
            v16bf b0 = load_frag_b_sw(wf + (size_t)(kt * 16 + nt0) * 512);
            v16bf b1 = load_frag_b_sw(wf + (size_t)(kt * 16 + nt0 + 1) * 512);
            acc0 = wmma_bf16(av, b0, acc0);
            acc1 = wmma_bf16(av, b1, acc1);
        }
        int n0 = nt0 * 16;
#pragma unroll
        for (int r = 0; r < 8; ++r) {
            int row = r0 + mt * 16 + mb + r;
            out[(size_t)row * CC + n0 + nn]      = acc0[r] + bias[n0 + nn];
            out[(size_t)row * CC + n0 + 16 + nn] = acc1[r] + bias[n0 + 16 + nn];
        }
    }
}

// ---------------------------------------------------------------------------
// Host launcher
// ---------------------------------------------------------------------------
extern "C" void kernel_launch(void* const* d_in, const int* in_sizes, int n_in,
                              void* d_out, int out_size, void* d_ws, size_t ws_size,
                              hipStream_t stream) {
    (void)in_sizes; (void)n_in; (void)out_size; (void)ws_size;
    const float* f1 = (const float*)d_in[0];
    const float* f2 = (const float*)d_in[1];
    // per-projection params: dw_w, dw_b, bn_g, bn_b, bn_m, bn_v, pw_w, pw_b
    const float* qp[8]; const float* kp[8]; const float* vp[8];
    for (int i = 0; i < 8; ++i) {
        qp[i] = (const float*)d_in[2 + i];
        kp[i] = (const float*)d_in[10 + i];
        vp[i] = (const float*)d_in[18 + i];
    }
    const float* ffn_w = (const float*)d_in[26];
    const float* ffn_b = (const float*)d_in[27];

    char* ws = (char*)d_ws;
    unsigned short* wq   = (unsigned short*)(ws);                      // 256*512*2
    unsigned short* wk   = (unsigned short*)(ws + 262144);
    unsigned short* wv   = (unsigned short*)(ws + 524288);
    unsigned short* wf   = (unsigned short*)(ws + 786432);             // 512*256*2
    unsigned short* q_ws = (unsigned short*)(ws + 1048576);            // 8 MB each
    unsigned short* k_ws = (unsigned short*)(ws + 1048576 + 8388608);
    unsigned short* v_ws = (unsigned short*)(ws + 1048576 + 2 * 8388608);
    unsigned short* a_ws = (unsigned short*)(ws + 1048576 + 3 * 8388608);

    prep_weights_kernel<<<512, 256, 0, stream>>>(qp[6], kp[6], vp[6], ffn_w,
                                                 wq, wk, wv, wf);

    dim3 gA(HWN / 64, BB);
    sepconv_proj_kernel<<<gA, 256, 0, stream>>>(f1, qp[0], qp[1], qp[2], qp[3],
        qp[4], qp[5], wq, qp[7], q_ws, SCALEV, 0);
    sepconv_proj_kernel<<<gA, 256, 0, stream>>>(f2, kp[0], kp[1], kp[2], kp[3],
        kp[4], kp[5], wk, kp[7], k_ws, 1.0f, 1);
    sepconv_proj_kernel<<<gA, 256, 0, stream>>>(f2, vp[0], vp[1], vp[2], vp[3],
        vp[4], vp[5], wv, vp[7], v_ws, 1.0f, 2);

    attention_kernel<<<dim3(8, HEADS, BB), 256, 0, stream>>>(q_ws, k_ws, v_ws, a_ws);

    ffn_kernel<<<dim3(128), 256, 0, stream>>>(a_ws, wf, ffn_b, (float*)d_out);
}
